// MDTA_16595753632433
// MI455X (gfx1250) — matrix operand order repaired and checked by reading, not compiled, not run on hardware
//
#include <hip/hip_runtime.h>

// ---------------- types / helpers ----------------
typedef __bf16 v8bf  __attribute__((ext_vector_type(8)));
typedef __bf16 v16bf __attribute__((ext_vector_type(16)));
typedef float  v8f   __attribute__((ext_vector_type(8)));

__device__ __forceinline__ unsigned short f2bf(float f) {
    unsigned int u = __builtin_bit_cast(unsigned int, f);
    u += 0x7FFFu + ((u >> 16) & 1u);        // round-to-nearest-even
    return (unsigned short)(u >> 16);
}
__device__ __forceinline__ float bf2f(unsigned short h) {
    unsigned int u = ((unsigned int)h) << 16;
    return __builtin_bit_cast(float, u);
}

__device__ __forceinline__ void gprefetch(const void* p) {
    __builtin_prefetch(p, 0, 3);            // lowers to global_prefetch_b8
}

union Frag { v16bf v; v8bf h[2]; };

// Build a 16-element bf16 fragment from two 8-element (16B) chunks.
__device__ __forceinline__ v16bf make_frag(const unsigned short* p0,
                                           const unsigned short* p1) {
    Frag f;
    f.h[0] = *(const v8bf*)p0;
    f.h[1] = *(const v8bf*)p1;
    return f.v;
}

__device__ __forceinline__ v8f wmma_bf16(v16bf a, v16bf b, v8f c) {
    // (neg_a, A, neg_b, B, c_mod, C, reuse_a, reuse_b)
    return __builtin_amdgcn_wmma_f32_16x16x32_bf16(false, a, false, b,
                                                   (short)0, c, false, false);
}

// ---------------- generic WMMA GEMM: C[M x 16384] = A[M x 192] * B[192 x 16384]
// A fp32 row-major (lda=192, optional per-batch stride). B fp32 or bf16.
// Block = 128 threads (4 waves); block tile 64M x 64N; wave tile 16M x 64N.
template<bool B_BF16, bool OUT_F32>
__global__ void __launch_bounds__(128)
gemm_wmma_kernel(const float* __restrict__ A, long aBatch,
                 const void* __restrict__ Bsrc, long bBatch,
                 void* __restrict__ Out, long oBatch)
{
    __shared__ __align__(16) unsigned short sA[64 * 40];
    __shared__ __align__(16) unsigned short sB[64 * 40];

    const int b  = blockIdx.z;
    const int m0 = blockIdx.y * 64;
    const int n0 = blockIdx.x * 64;
    const int t = threadIdx.x, lane = t & 31, w = t >> 5;
    const int half = lane >> 4, l16 = lane & 15;

    const float*          Ab  = A + (size_t)b * aBatch;
    const unsigned short* B16 = (const unsigned short*)Bsrc + (size_t)b * bBatch;
    const float*          B32 = (const float*)Bsrc + (size_t)b * bBatch;

    v8f acc[4] = {};

    for (int k0 = 0; k0 < 192; k0 += 32) {
        // stage A tile 64x32 (bf16, padded stride 40)
        #pragma unroll
        for (int i = 0; i < 16; ++i) {
            int idx = t + i * 128;
            int r = idx >> 5, c = idx & 31;
            sA[r * 40 + c] = f2bf(Ab[(size_t)(m0 + r) * 192 + (k0 + c)]);
        }
        // stage B^T tile: sB[n][k] = B[k0+k][n0+n]   (coalesced global reads)
        #pragma unroll
        for (int i = 0; i < 16; ++i) {
            int idx = t + i * 128;
            int kk = idx >> 6, n = idx & 63;
            size_t g = (size_t)(k0 + kk) * 16384 + (size_t)(n0 + n);
            sB[n * 40 + kk] = B_BF16 ? B16[g] : f2bf(B32[g]);
        }
        // prefetch next K-step tiles while this step computes
        if (k0 + 32 < 192) {
            {   // next B tile: 32 rows x 64 cols
                int r = t & 31;
                size_t g = (size_t)(k0 + 32 + r) * 16384 + (size_t)n0 +
                           (size_t)((t >> 5) << 4);
                if (B_BF16) gprefetch(B16 + g);
                else        gprefetch(B32 + g);
            }
            {   // next A tile: 64 rows x 32 cols
                int r = t >> 1;
                gprefetch(&Ab[(size_t)(m0 + r) * 192 + (k0 + 32) + (t & 1) * 16]);
            }
        }
        __syncthreads();

        const int arow = w * 16 + l16;
        v16bf af = make_frag(&sA[arow * 40 + half * 8],
                             &sA[arow * 40 + 16 + half * 8]);
        #pragma unroll
        for (int nt = 0; nt < 4; ++nt) {
            const int brow = nt * 16 + l16;
            v16bf bf = make_frag(&sB[brow * 40 + half * 8],
                                 &sB[brow * 40 + 16 + half * 8]);
            acc[nt] = wmma_bf16(af, bf, acc[nt]);
        }
        __syncthreads();
    }

    // D layout: elem r -> M = r + 8*half, N = lane&15
    #pragma unroll
    for (int nt = 0; nt < 4; ++nt) {
        #pragma unroll
        for (int r = 0; r < 8; ++r) {
            int row = m0 + w * 16 + r + 8 * half;
            int col = n0 + nt * 16 + l16;
            size_t g = (size_t)row * 16384 + col;
            if (OUT_F32)
                ((float*)Out + (size_t)b * oBatch)[g] = acc[nt][r];
            else
                ((unsigned short*)Out + (size_t)b * oBatch)[g] = f2bf(acc[nt][r]);
        }
    }
}

// ---------------- depthwise 3x3 conv (pad=1) + fused sum-of-squares for q,k
__global__ void __launch_bounds__(256)
dwconv_kernel(const unsigned short* __restrict__ qkv0,
              const float* __restrict__ w_dw,
              unsigned short* __restrict__ qkv_dw,
              float* __restrict__ norms)
{
    __shared__ float tile[10][34];
    __shared__ float red[256];
    const int b = blockIdx.z, c = blockIdx.y;
    const int x0 = (blockIdx.x & 3) * 32;
    const int y0 = (blockIdx.x >> 2) * 8;
    const int t = threadIdx.x;
    const int tx = t & 31, ty = t >> 5;

    const unsigned short* in = qkv0 + ((size_t)b * 576 + c) * 16384;

    for (int idx = t; idx < 340; idx += 256) {
        int ry = idx / 34, cx = idx - ry * 34;
        int gy = y0 - 1 + ry, gx = x0 - 1 + cx;
        float v = 0.f;
        if (gy >= 0 && gy < 128 && gx >= 0 && gx < 128)
            v = bf2f(in[gy * 128 + gx]);
        tile[ry][cx] = v;
    }
    float wv[9];
    #pragma unroll
    for (int j = 0; j < 9; ++j) wv[j] = w_dw[c * 9 + j];
    __syncthreads();

    float y = 0.f;
    #pragma unroll
    for (int ky = 0; ky < 3; ++ky)
        #pragma unroll
        for (int kx = 0; kx < 3; ++kx)
            y = fmaf(wv[ky * 3 + kx], tile[ty + ky][tx + kx], y);

    qkv_dw[((size_t)b * 576 + c) * 16384 + (y0 + ty) * 128 + (x0 + tx)] = f2bf(y);

    if (c < 384) {                       // q,k channels: accumulate ||.||^2 over HW
        red[t] = y * y;
        __syncthreads();
        for (int s = 128; s > 0; s >>= 1) {
            if (t < s) red[t] += red[t + s];
            __syncthreads();
        }
        if (t == 0) atomicAdd(&norms[b * 384 + c], red[0]);
    }
}

// ---------------- scores = (q/||q||)(k/||k||)^T * temp ; softmax -> attn
// One block (8 waves) per (b, head). Each wave owns full 48x48, disjoint K slices.
__global__ void __launch_bounds__(256)
scores_kernel(const unsigned short* __restrict__ qkv_dw,
              const float* __restrict__ norms,
              const float* __restrict__ temperature,
              float* __restrict__ attn)
{
    __shared__ float s_sc[48 * 48];
    __shared__ float s_qn[48], s_kn[48];
    const int head = blockIdx.x, b = blockIdx.y;
    const int t = threadIdx.x, lane = t & 31, wid = t >> 5;
    const int half = lane >> 4, l16 = lane & 15;

    for (int i = t; i < 48 * 48; i += 256) s_sc[i] = 0.f;
    if (t < 48)
        s_qn[t] = fmaxf(sqrtf(norms[b * 384 + head * 48 + t]), 1e-12f);
    else if (t >= 64 && t < 112) {
        int e = t - 64;
        s_kn[e] = fmaxf(sqrtf(norms[b * 384 + 192 + head * 48 + e]), 1e-12f);
    }
    __syncthreads();

    const unsigned short* qb = qkv_dw + ((size_t)b * 576 + head * 48) * 16384;
    const unsigned short* kb = qkv_dw + ((size_t)b * 576 + 192 + head * 48) * 16384;

    v8f acc[3][3] = {};
    for (int step = wid; step < 512; step += 8) {
        const int k = step * 32;
        v16bf af[3], bfg[3];
        #pragma unroll
        for (int mt = 0; mt < 3; ++mt) {
            const unsigned short* p = qb + (size_t)(mt * 16 + l16) * 16384 + k;
            af[mt] = make_frag(p + half * 8, p + 16 + half * 8);
        }
        #pragma unroll
        for (int nt = 0; nt < 3; ++nt) {
            const unsigned short* p =
                kb + (size_t)(nt * 16 + l16) * 16384 + k;
            bfg[nt] = make_frag(p + half * 8, p + 16 + half * 8);
        }
        // prefetch this wave's next K slice (k + 256 elements = +512 B per row)
        const int kn = k + 256;
        if (kn < 16384) {
            #pragma unroll
            for (int mt = 0; mt < 3; ++mt)
                gprefetch(qb + (size_t)(mt * 16 + l16) * 16384 + kn + half * 8);
            #pragma unroll
            for (int nt = 0; nt < 3; ++nt)
                gprefetch(kb + (size_t)(nt * 16 + l16) * 16384 + kn + half * 16);
        }
        #pragma unroll
        for (int mt = 0; mt < 3; ++mt)
            #pragma unroll
            for (int nt = 0; nt < 3; ++nt)
                acc[mt][nt] = wmma_bf16(af[mt], bfg[nt], acc[mt][nt]);
    }
    // cross-wave reduction into LDS
    #pragma unroll
    for (int mt = 0; mt < 3; ++mt)
        #pragma unroll
        for (int nt = 0; nt < 3; ++nt)
            #pragma unroll
            for (int r = 0; r < 8; ++r) {
                int d = mt * 16 + r + 8 * half;
                int e = nt * 16 + l16;
                atomicAdd(&s_sc[d * 48 + e], acc[mt][nt][r]);
            }
    __syncthreads();

    if (t < 48) {                          // row-wise softmax
        const int d = t;
        const float tmp = temperature[head];
        const float qn = s_qn[d];
        float row[48];
        float mx = -3.4e38f;
        for (int e = 0; e < 48; ++e) {
            float v = s_sc[d * 48 + e] * tmp / (qn * s_kn[e]);
            row[e] = v;
            mx = fmaxf(mx, v);
        }
        float sum = 0.f;
        for (int e = 0; e < 48; ++e) {
            float ex = __expf(row[e] - mx);
            row[e] = ex;
            sum += ex;
        }
        float inv = 1.f / sum;
        float* outp = attn + (((size_t)(b * 4 + head) * 48) + d) * 48;
        for (int e = 0; e < 48; ++e) outp[e] = row[e] * inv;
    }
}

// ---------------- P[b][o][h*48+e] = sum_d w_out[o][h*48+d] * attn[b][h][d][e]
// (fuses the output 1x1 conv with attn so that out = P @ v is a single GEMM)
__global__ void __launch_bounds__(256)
pmat_kernel(const float* __restrict__ w_out,
            const float* __restrict__ attn,
            float* __restrict__ P)
{
    int idx = blockIdx.x * 256 + threadIdx.x;
    if (idx >= 8 * 192 * 192) return;
    int b = idx / (192 * 192);
    int rem = idx - b * 192 * 192;
    int o = rem / 192;
    int col = rem - o * 192;
    int h = col / 48, e = col - h * 48;
    const float* wrow = w_out + o * 192 + h * 48;
    const float* arow = attn + ((size_t)(b * 4 + h) * 48) * 48 + e;
    float s = 0.f;
    #pragma unroll 8
    for (int d = 0; d < 48; ++d) s = fmaf(wrow[d], arow[d * 48], s);
    P[idx] = s;
}

// ---------------- host-side launch ----------------
extern "C" void kernel_launch(void* const* d_in, const int* in_sizes, int n_in,
                              void* d_out, int out_size, void* d_ws, size_t ws_size,
                              hipStream_t stream)
{
    (void)in_sizes; (void)n_in; (void)out_size; (void)ws_size;
    const float* x           = (const float*)d_in[0];  // [8,192,128,128]
    const float* w_qkv       = (const float*)d_in[1];  // [576,192]
    const float* w_dw        = (const float*)d_in[2];  // [576,1,3,3]
    const float* temperature = (const float*)d_in[3];  // [1,4,1,1]
    const float* w_out       = (const float*)d_in[4];  // [192,192]
    float* out = (float*)d_out;                        // [8,192,128,128] fp32

    char* ws = (char*)d_ws;
    unsigned short* qkv0  = (unsigned short*)ws;                    // 150,994,944 B (bf16)
    unsigned short* qkvdw = (unsigned short*)(ws + 150994944ull);   // 150,994,944 B (bf16)
    float* norms = (float*)(ws + 301989888ull);                     //      12,288 B
    float* attn  = (float*)(ws + 302002176ull);                     //   1,179,648 B
    float* P     = (float*)(ws + 303181824ull);                     //   1,179,648 B

    hipMemsetAsync(norms, 0, 8 * 384 * sizeof(float), stream);

    // 1) qkv = w_qkv @ x     (M=576, K=192, N=16384 per batch), bf16 output
    gemm_wmma_kernel<false, false><<<dim3(256, 9, 8), 128, 0, stream>>>(
        w_qkv, 0L, (const void*)x, (long)192 * 16384,
        (void*)qkv0, (long)576 * 16384);

    // 2) depthwise 3x3 conv + fused ||q||^2, ||k||^2 reductions
    dwconv_kernel<<<dim3(64, 576, 8), 256, 0, stream>>>(qkv0, w_dw, qkvdw, norms);

    // 3) attention scores + softmax per (b, head)
    scores_kernel<<<dim3(4, 8, 1), 256, 0, stream>>>(qkvdw, norms, temperature, attn);

    // 4) fold w_out into attn -> P
    pmat_kernel<<<dim3(1152), 256, 0, stream>>>(w_out, attn, P);

    // 5) out = P @ v          (M=192, K=192, N=16384 per batch), fp32 output
    gemm_wmma_kernel<true, true><<<dim3(256, 3, 8), 128, 0, stream>>>(
        P, (long)192 * 192,
        (const void*)(qkvdw + (size_t)384 * 16384), (long)576 * 16384,
        (void*)out, (long)192 * 16384);
}
